// GAT_70772471103692
// MI455X (gfx1250) — matrix operand order repaired and checked by reading, not compiled, not run on hardware
//
#include <hip/hip_runtime.h>

// ---------------------------------------------------------------------------
// GAT(256 -> 2x64, concat) + GCN(128 -> 64) + log_softmax for gfx1250 (MI455X)
// GEMMs via v_wmma_f32_16x16x32_f16 (wave32 WMMA), scatters via f32 atomics.
// ---------------------------------------------------------------------------

typedef __attribute__((ext_vector_type(16))) _Float16 v16h;
typedef __attribute__((ext_vector_type(8)))  _Float16 v8h;
typedef __attribute__((ext_vector_type(8)))  float    v8f;

#define F_IN 256
#define HC   128   // H*C
#define NC   64

// ---- order-preserving float<->uint encoding for atomic max --------------
__device__ __forceinline__ unsigned encF(float f) {
  unsigned u = __float_as_uint(f);
  return (u & 0x80000000u) ? ~u : (u | 0x80000000u);
}
__device__ __forceinline__ float decF(unsigned k) {
  unsigned u = (k & 0x80000000u) ? (k & 0x7FFFFFFFu) : ~k;
  return __uint_as_float(u);
}

// ---- generic zero fill ---------------------------------------------------
__global__ void k_zero(float* __restrict__ p, long n) {
  long g = (long)blockIdx.x * blockDim.x + threadIdx.x;
  long stride = (long)gridDim.x * blockDim.x;
  for (; g < n; g += stride) p[g] = 0.0f;
}

// ---- convert x (f32, [N,256]) -> xh (f16, [Npad,256], zero-padded) -------
__global__ void k_convert_x(const float* __restrict__ x, _Float16* __restrict__ xh,
                            long n_valid, long n_total) {
  long g = (long)blockIdx.x * blockDim.x + threadIdx.x;
  if (g >= n_total) return;
  xh[g] = (_Float16)((g < n_valid) ? x[g] : 0.0f);
}

// ---- transpose+convert weights: W1[256,128]->w1t[128,256], W2[128,64]->w2t[64,128]
__global__ void k_prep_w(const float* __restrict__ W1, const float* __restrict__ W2,
                         _Float16* __restrict__ w1t, _Float16* __restrict__ w2t) {
  int stride = gridDim.x * blockDim.x;
  for (int i = blockIdx.x * blockDim.x + threadIdx.x; i < F_IN * HC; i += stride) {
    int k = i / HC, n = i % HC;
    w1t[(size_t)n * F_IN + k] = (_Float16)W1[i];
  }
  for (int i = blockIdx.x * blockDim.x + threadIdx.x; i < HC * NC; i += stride) {
    int k = i / NC, n = i % NC;
    w2t[(size_t)n * HC + k] = (_Float16)W2[i];
  }
}

// ---- GEMM1: h[Npad,128] = xh[Npad,256] @ W1 (w1t is [128,256] f16) -------
// block = 256 threads = 8 waves; wave w owns 16x16 output tile (tm, w).
__global__ void k_gemm1(const _Float16* __restrict__ xh, const _Float16* __restrict__ w1t,
                        float* __restrict__ h) {
  const int lane = threadIdx.x & 31;
  const int wave = threadIdx.x >> 5;          // n-tile 0..7
  const int tm   = blockIdx.x;                // m-tile
  const int row  = tm * 16 + (lane & 15);
  const int kA   = (lane >> 4) * 16;          // A K-split: 0 / 16
  const int ncol = wave * 16 + (lane & 15);
  const int kB   = (lane >> 4) * 16;          // B K-split: 0 / 16
  v8f acc = {};
#pragma unroll
  for (int k0 = 0; k0 < F_IN; k0 += 32) {
    v8h alo = *(const v8h*)(xh + (size_t)row * F_IN + k0 + kA);
    v8h ahi = *(const v8h*)(xh + (size_t)row * F_IN + k0 + kA + 8);
    v16h a;
#pragma unroll
    for (int i = 0; i < 8; ++i) { a[i] = alo[i]; a[i + 8] = ahi[i]; }
    v16h b = *(const v16h*)(w1t + (size_t)ncol * F_IN + k0 + kB);
    acc = __builtin_amdgcn_wmma_f32_16x16x32_f16(false, a, false, b,
                                                 (short)0, acc, false, false);
  }
  const int mbase = tm * 16 + (lane >> 4) * 8;
  const int col   = wave * 16 + (lane & 15);
#pragma unroll
  for (int v = 0; v < 8; ++v)
    h[(size_t)(mbase + v) * HC + col] = acc[v];
}

// ---- attention coefficients: a_src/a_dst [N,2] ---------------------------
__global__ void k_att(const float* __restrict__ h, const float* __restrict__ att_src,
                      const float* __restrict__ att_dst,
                      float* __restrict__ asrc, float* __restrict__ adst, int N) {
  int t = blockIdx.x * blockDim.x + threadIdx.x;
  if (t >= N * 2) return;
  int node = t >> 1, hh = t & 1;
  const float* hp = h + (size_t)node * HC + hh * 64;
  const float* as = att_src + hh * 64;
  const float* ad = att_dst + hh * 64;
  float s0 = 0.f, s1 = 0.f;
#pragma unroll 8
  for (int c = 0; c < 64; ++c) { float v = hp[c]; s0 += v * as[c]; s1 += v * ad[c]; }
  asrc[t] = s0;
  adst[t] = s1;
}

// ---- edge pass 1: leaky-relu logits, segment max (encoded), degree -------
__global__ void k_edge_max(const int* __restrict__ ei, const float* __restrict__ asrc,
                           const float* __restrict__ adst, unsigned* __restrict__ m,
                           float* __restrict__ deg, float* __restrict__ ebuf,
                           int E, int N) {
  int t = blockIdx.x * blockDim.x + threadIdx.x;
  if (t >= E + N) return;
  int s = (t < E) ? ei[t] : (t - E);
  int d = (t < E) ? ei[E + t] : (t - E);
#pragma unroll
  for (int hh = 0; hh < 2; ++hh) {
    float e = asrc[s * 2 + hh] + adst[d * 2 + hh];
    e = (e > 0.f) ? e : 0.2f * e;
    ebuf[(size_t)t * 2 + hh] = e;
    atomicMax(&m[d * 2 + hh], encF(e));
  }
  atomicAdd(&deg[d], 1.0f);
}

// ---- edge pass 2: p = exp(e - m[dst]), segment sum -----------------------
__global__ void k_edge_exp(const int* __restrict__ ei, const unsigned* __restrict__ m,
                           float* __restrict__ ebuf, float* __restrict__ ssum,
                           int E, int N) {
  int t = blockIdx.x * blockDim.x + threadIdx.x;
  if (t >= E + N) return;
  int d = (t < E) ? ei[E + t] : (t - E);
#pragma unroll
  for (int hh = 0; hh < 2; ++hh) {
    float p = __expf(ebuf[(size_t)t * 2 + hh] - decF(m[d * 2 + hh]));
    ebuf[(size_t)t * 2 + hh] = p;
    atomicAdd(&ssum[d * 2 + hh], p);
  }
}

// ---- edge pass 3: out[dst,:] += h[src,:] * alpha -------------------------
__global__ void k_edge_agg(const int* __restrict__ ei, const float* __restrict__ h,
                           const float* __restrict__ ebuf, const float* __restrict__ ssum,
                           float* __restrict__ out, int E, int N) {
  long g = (long)blockIdx.x * blockDim.x + threadIdx.x;
  long tot = (long)(E + N) * HC;
  if (g >= tot) return;
  int t = (int)(g >> 7);
  int c = (int)(g & 127);
  int s = (t < E) ? ei[t] : (t - E);
  int d = (t < E) ? ei[E + t] : (t - E);
  int hh = c >> 6;
  float alpha = ebuf[(size_t)t * 2 + hh] / (ssum[d * 2 + hh] + 1e-16f);
  atomicAdd(&out[(size_t)d * HC + c], h[(size_t)s * HC + c] * alpha);
}

// ---- bias + relu + convert to f16 ([Npad,128], zero pad) -----------------
__global__ void k_relu_conv(const float* __restrict__ out, const float* __restrict__ b1,
                            _Float16* __restrict__ outh, long n_valid, long n_total) {
  long g = (long)blockIdx.x * blockDim.x + threadIdx.x;
  if (g >= n_total) return;
  float v = 0.f;
  if (g < n_valid) {
    v = out[g] + b1[g & (HC - 1)];
    v = (v > 0.f) ? v : 0.f;
  }
  outh[g] = (_Float16)v;
}

// ---- GEMM2: z[Npad,64] = outh[Npad,128] @ W2 (w2t is [64,128] f16) -------
// block = 128 threads = 4 waves; wave w owns 16x16 output tile (tm, w).
__global__ void k_gemm2(const _Float16* __restrict__ outh, const _Float16* __restrict__ w2t,
                        float* __restrict__ z) {
  const int lane = threadIdx.x & 31;
  const int wave = threadIdx.x >> 5;          // n-tile 0..3
  const int tm   = blockIdx.x;
  const int row  = tm * 16 + (lane & 15);
  const int kA   = (lane >> 4) * 16;
  const int ncol = wave * 16 + (lane & 15);
  const int kB   = (lane >> 4) * 16;
  v8f acc = {};
#pragma unroll
  for (int k0 = 0; k0 < HC; k0 += 32) {
    v8h alo = *(const v8h*)(outh + (size_t)row * HC + k0 + kA);
    v8h ahi = *(const v8h*)(outh + (size_t)row * HC + k0 + kA + 8);
    v16h a;
#pragma unroll
    for (int i = 0; i < 8; ++i) { a[i] = alo[i]; a[i + 8] = ahi[i]; }
    v16h b = *(const v16h*)(w2t + (size_t)ncol * HC + k0 + kB);
    acc = __builtin_amdgcn_wmma_f32_16x16x32_f16(false, a, false, b,
                                                 (short)0, acc, false, false);
  }
  const int mbase = tm * 16 + (lane >> 4) * 8;
  const int col   = wave * 16 + (lane & 15);
#pragma unroll
  for (int v = 0; v < 8; ++v)
    z[(size_t)(mbase + v) * NC + col] = acc[v];
}

// ---- GCN scatter: dout[dst,:] += z[src,:] * dinv[src]*dinv[dst] ----------
__global__ void k_edge_gcn(const int* __restrict__ ei, const float* __restrict__ z,
                           const float* __restrict__ deg, float* __restrict__ dout,
                           int E, int N) {
  long g = (long)blockIdx.x * blockDim.x + threadIdx.x;
  long tot = (long)(E + N) * NC;
  if (g >= tot) return;
  int t = (int)(g >> 6);
  int c = (int)(g & 63);
  int s = (t < E) ? ei[t] : (t - E);
  int d = (t < E) ? ei[E + t] : (t - E);
  float nrm = rsqrtf(deg[s]) * rsqrtf(deg[d]);
  atomicAdd(&dout[(size_t)d * NC + c], z[(size_t)s * NC + c] * nrm);
}

// ---- + b2, then row-wise log_softmax (one wave32 per node) ---------------
__global__ void k_lsm(float* __restrict__ dout, const float* __restrict__ b2, int N) {
  const int lane = threadIdx.x & 31;
  const int wave = threadIdx.x >> 5;
  const int node = blockIdx.x * (blockDim.x >> 5) + wave;
  if (node >= N) return;
  float v0 = dout[(size_t)node * NC + lane]      + b2[lane];
  float v1 = dout[(size_t)node * NC + lane + 32] + b2[lane + 32];
  float mx = fmaxf(v0, v1);
#pragma unroll
  for (int o = 16; o >= 1; o >>= 1) mx = fmaxf(mx, __shfl_xor(mx, o, 32));
  float se = __expf(v0 - mx) + __expf(v1 - mx);
#pragma unroll
  for (int o = 16; o >= 1; o >>= 1) se += __shfl_xor(se, o, 32);
  float l = mx + __logf(se);
  dout[(size_t)node * NC + lane]      = v0 - l;
  dout[(size_t)node * NC + lane + 32] = v1 - l;
}

// ---------------------------------------------------------------------------
extern "C" void kernel_launch(void* const* d_in, const int* in_sizes, int n_in,
                              void* d_out, int out_size, void* d_ws, size_t ws_size,
                              hipStream_t stream) {
  (void)n_in; (void)out_size; (void)ws_size;
  const float* x       = (const float*)d_in[0];
  const int*   ei      = (const int*)  d_in[1];
  const float* W1      = (const float*)d_in[2];
  const float* att_src = (const float*)d_in[3];
  const float* att_dst = (const float*)d_in[4];
  const float* b1      = (const float*)d_in[5];
  const float* W2      = (const float*)d_in[6];
  const float* b2      = (const float*)d_in[7];
  float* dout = (float*)d_out;

  const int N    = in_sizes[0] / F_IN;
  const int E    = in_sizes[1] / 2;
  const int EP   = E + N;
  const int Npad = ((N + 15) / 16) * 16;
  const int Mt   = Npad / 16;

  // ---- workspace layout (256B aligned regions) ----
  char* ws = (char*)d_ws;
  size_t off = 0;
  auto alloc = [&](size_t bytes) -> void* {
    void* p = ws + off;
    off = (off + bytes + 255) & ~(size_t)255;
    return p;
  };
  _Float16* xh   = (_Float16*)alloc((size_t)Npad * F_IN * 2);  // dead after GEMM1
  _Float16* w1t  = (_Float16*)alloc((size_t)HC * F_IN * 2);
  _Float16* w2t  = (_Float16*)alloc((size_t)NC * HC * 2);
  float*    hbuf = (float*)   alloc((size_t)Npad * HC * 4);
  float*    asrc = (float*)   alloc((size_t)N * 2 * 4);
  float*    adst = (float*)   alloc((size_t)N * 2 * 4);
  unsigned* mmax = (unsigned*)alloc((size_t)N * 2 * 4);
  float*    ssum = (float*)   alloc((size_t)N * 2 * 4);
  float*    deg  = (float*)   alloc((size_t)N * 4);
  float*    ebuf = (float*)   alloc((size_t)EP * 2 * 4);
  float*    outf = (float*)   alloc((size_t)Npad * HC * 4);
  // reuse xh region: outh (Npad*128 f16) + z (Npad*64 f32) == Npad*256 f16
  _Float16* outh = xh;
  float*    z    = (float*)((char*)xh + (size_t)Npad * HC * 2);

  const int B = 256;
  auto cdiv = [](long a, long b) { return (int)((a + b - 1) / b); };

  // zero accumulators (every call; output/ws are poisoned once by harness)
  k_zero<<<cdiv((long)Npad * HC, B), B, 0, stream>>>(outf, (long)Npad * HC);
  k_zero<<<cdiv(N, B), B, 0, stream>>>(deg, N);
  k_zero<<<cdiv(N * 2, B), B, 0, stream>>>(ssum, N * 2);
  k_zero<<<cdiv(N * 2, B), B, 0, stream>>>((float*)mmax, N * 2);  // 0u = -inf key
  k_zero<<<cdiv((long)N * NC, B), B, 0, stream>>>(dout, (long)N * NC);

  // stage f16 operands
  k_convert_x<<<cdiv((long)Npad * F_IN, B), B, 0, stream>>>(x, xh, (long)N * F_IN,
                                                            (long)Npad * F_IN);
  k_prep_w<<<64, B, 0, stream>>>(W1, W2, w1t, w2t);

  // GAT layer
  k_gemm1<<<Mt, 256, 0, stream>>>(xh, w1t, hbuf);
  k_att<<<cdiv(N * 2, B), B, 0, stream>>>(hbuf, att_src, att_dst, asrc, adst, N);
  k_edge_max<<<cdiv(EP, B), B, 0, stream>>>(ei, asrc, adst, mmax, deg, ebuf, E, N);
  k_edge_exp<<<cdiv(EP, B), B, 0, stream>>>(ei, mmax, ebuf, ssum, E, N);
  k_edge_agg<<<cdiv((long)EP * HC, B), B, 0, stream>>>(ei, hbuf, ebuf, ssum, outf, E, N);
  k_relu_conv<<<cdiv((long)Npad * HC, B), B, 0, stream>>>(outf, b1, outh,
                                                          (long)N * HC, (long)Npad * HC);

  // GCN layer + log_softmax
  k_gemm2<<<Mt, 128, 0, stream>>>(outh, w2t, z);
  k_edge_gcn<<<cdiv((long)EP * NC, B), B, 0, stream>>>(ei, z, deg, dout, E, N);
  k_lsm<<<cdiv(N, 8), 256, 0, stream>>>(dout, b2, N);
}
